// SNNEncoder_44530220925425
// MI455X (gfx1250) — compile-verified, matrix operand order checked by reading, not run on hardware
//
#include <hip/hip_runtime.h>

// ---------------------------------------------------------------------------
// SNN encoder for MI455X (gfx1250, wave32, WMMA).
//
// Stage P (prep): pack W0/W1/W2 into WMMA B-fragment order as f16 hi/lo pairs
//                 (w = hi + lo, error ~2^-22); fold bias+BN into alpha/beta.
// Stage A:        H0 = BN0(x @ W0^T + b0) once (x is identical every step).
//                 f16 emulated-fp32 GEMM: acc += xh*wh + xl*wh + xh*wl.
// Stage B:        persistent temporal kernel; 256 blocks x 64 batch rows,
//                 T-loop inside. m0 in global (L2-resident), m1/m2/rate in
//                 registers, spikes (exact in f16) staged through padded LDS.
//                 spike_rec written with non-temporal stores (write-once).
// ---------------------------------------------------------------------------

typedef _Float16 half_t;
typedef __attribute__((ext_vector_type(16))) _Float16 v16h;
typedef __attribute__((ext_vector_type(8)))  float    v8f;

#define WMMA_F16(a, b, c) \
  __builtin_amdgcn_wmma_f32_16x16x32_f16(false, (a), false, (b), (short)0, (c), false, false)

// 16-bit A-matrix 16x32 per-lane K mapping (ISA 7.12.2):
// lane row m = lane&15, half h = lane>>4; element e -> k:
//   e 0..7  -> K =      h*8 + e      (VGPR0-3)
//   e 8..15 -> K = 16 + h*8 + (e-8)  (VGPR4-7)
__device__ __forceinline__ int frag_k(int e, int lane) {
  return ((e >> 3) << 4) + ((lane >> 4) << 3) + (e & 7);
}

// Gather an A-fragment from a row-major LDS tile (stride ld halves).
__device__ __forceinline__ v16h load_a_frag_lds(const half_t* base, int ld, int kbase, int lane) {
  const half_t* row = base + (lane & 15) * ld;
  v16h a;
#pragma unroll
  for (int e = 0; e < 16; ++e) a[e] = row[kbase + frag_k(e, lane)];
  return a;
}

// B-fragments are pre-packed [ntile][kc][lane][16] -> one contiguous 32B load.
__device__ __forceinline__ v16h load_b_frag(const half_t* p, int ntile, int kc, int nk, int lane) {
  return *reinterpret_cast<const v16h*>(p + ((((size_t)ntile * nk + kc) * 32 + lane) << 4));
}

// --------------------------- prep kernels ----------------------------------

// Pack W[fo][fi] (row-major) into B-fragment order, f16 hi/lo split.
// B element (k, n) = W[n][k].  Layout index: [ntile][kc][lane][e].
__global__ void snn_pack_weights(const float* __restrict__ W,
                                 half_t* __restrict__ hi, half_t* __restrict__ lo,
                                 int fo, int fi) {
  int idx = blockIdx.x * blockDim.x + threadIdx.x;
  if (idx >= fo * fi) return;
  int e    = idx & 15;
  int lane = (idx >> 4) & 31;
  int nk   = fi >> 5;
  int tmp  = idx >> 9;
  int kc   = tmp % nk;
  int ntl  = tmp / nk;
  int n = ntl * 16 + (lane & 15);
  int k = kc * 32 + frag_k(e, lane);
  float w = W[(size_t)n * fi + k];
  half_t h = (half_t)w;
  hi[idx] = h;
  lo[idx] = (half_t)(w - (float)h);
}

// alpha = gamma * rsqrt(rvar+eps); beta = (b - rmean)*alpha + betabn
__global__ void snn_prep_bn(const float* __restrict__ b, const float* __restrict__ gamma,
                            const float* __restrict__ betabn, const float* __restrict__ rmean,
                            const float* __restrict__ rvar,
                            float* __restrict__ alpha, float* __restrict__ beta, int n) {
  int i = blockIdx.x * blockDim.x + threadIdx.x;
  if (i >= n) return;
  float a = gamma[i] * rsqrtf(rvar[i] + 1e-5f);
  alpha[i] = a;
  beta[i]  = (b[i] - rmean[i]) * a + betabn[i];
}

// --------------------------- stage A: H0 GEMM ------------------------------
// Block: 32 rows x all 256 out-channels. 8 waves x 4 tiles (2 tm x 16 tn).
// x split into f16 hi/lo on the fly; 3 WMMAs per K32-chunk per tile.
__global__ void __launch_bounds__(256) snn_h0_gemm(
    const float* __restrict__ x,                     // [16384,1024]
    const half_t* __restrict__ bhi, const half_t* __restrict__ blo,  // packed W0
    const float* __restrict__ alpha, const float* __restrict__ beta, // [256]
    float* __restrict__ H0)                          // [16384,256]
{
  __shared__ float xs[32 * 33];                      // padded fp32 x chunk
  const int tid = threadIdx.x, wave = tid >> 5, lane = tid & 31;
  const int rowbase = blockIdx.x * 32;
  const int tm  = wave >> 2;                         // 0..1
  const int tnb = (wave & 3) * 4;                    // 0,4,8,12

  v8f acc[4] = {};
  for (int kc = 0; kc < 32; ++kc) {
    __syncthreads();
#pragma unroll
    for (int j = 0; j < 4; ++j) {                    // stage 32x32 fp32 chunk
      int f = j * 256 + tid, r = f >> 5, c = f & 31;
      xs[r * 33 + c] = x[(size_t)(rowbase + r) * 1024 + (size_t)kc * 32 + c];
    }
    __syncthreads();

    const float* row = xs + (tm * 16 + (lane & 15)) * 33;
    v16h ah, al;
#pragma unroll
    for (int e = 0; e < 16; ++e) {                   // split x -> hi/lo f16
      float v = row[frag_k(e, lane)];
      half_t h = (half_t)v;
      ah[e] = h;
      al[e] = (half_t)(v - (float)h);
    }
#pragma unroll
    for (int i = 0; i < 4; ++i) {
      v16h bh = load_b_frag(bhi, tnb + i, kc, 32, lane);
      v16h bl = load_b_frag(blo, tnb + i, kc, 32, lane);
      acc[i] = WMMA_F16(ah, bh, acc[i]);
      acc[i] = WMMA_F16(al, bh, acc[i]);
      acc[i] = WMMA_F16(ah, bl, acc[i]);
    }
  }
  // C layout: lane n = lane&15; VGPR r -> m = r + 8*(lane>>4)
#pragma unroll
  for (int i = 0; i < 4; ++i) {
    int n = (tnb + i) * 16 + (lane & 15);
    float a = alpha[n], be = beta[n];
#pragma unroll
    for (int r = 0; r < 8; ++r) {
      int m = tm * 16 + r + ((lane >> 4) << 3);
      H0[(size_t)(rowbase + m) * 256 + n] = acc[i][r] * a + be;
    }
  }
}

// --------------------------- stage B: temporal loop ------------------------
__global__ void __launch_bounds__(256) snn_temporal(
    const float* __restrict__ H0,                    // [16384,256]
    float* __restrict__ m0g,                         // [16384,256] ws
    const half_t* __restrict__ b1hi, const half_t* __restrict__ b1lo,
    const half_t* __restrict__ b2hi, const half_t* __restrict__ b2lo,
    const float* __restrict__ alpha1, const float* __restrict__ beta1,
    const float* __restrict__ alpha2, const float* __restrict__ beta2,
    const float* __restrict__ lb0p, const float* __restrict__ th0p,
    const float* __restrict__ lb1p, const float* __restrict__ th1p,
    const float* __restrict__ lb2p, const float* __restrict__ th2p,
    const int* __restrict__ Tptr,
    float* __restrict__ out_rate,                    // [16384,64]
    float* __restrict__ out_rec)                     // [T,16384,64]
{
  __shared__ half_t s0[64 * 264];                    // spikes0, padded stride
  __shared__ half_t s1[64 * 136];                    // spikes1, padded stride
  const int tid = threadIdx.x, wave = tid >> 5, lane = tid & 31;
  const int rowbase = blockIdx.x * 64;
  const int T = *Tptr;
  const float lb0 = fminf(fmaxf(*lb0p, 0.f), 1.f), th0 = *th0p;
  const float lb1 = fminf(fmaxf(*lb1p, 0.f), 1.f), th1 = *th1p;
  const float lb2 = fminf(fmaxf(*lb2p, 0.f), 1.f), th2 = *th2p;

  // Stable tile ownership across t (lets membranes live in registers).
  const int tm1 = wave >> 1, tn1b = (wave & 1) * 4;  // layer1: 4x8 tiles
  const int tm2 = wave >> 1, tn2b = (wave & 1) * 2;  // layer2: 4x4 tiles

  float al1[4], be1[4], al2[2], be2[2];
#pragma unroll
  for (int i = 0; i < 4; ++i) { int n = (tn1b + i) * 16 + (lane & 15); al1[i] = alpha1[n]; be1[i] = beta1[n]; }
#pragma unroll
  for (int i = 0; i < 2; ++i) { int n = (tn2b + i) * 16 + (lane & 15); al2[i] = alpha2[n]; be2[i] = beta2[n]; }

  float m1[4][8], m2[2][8], rate[2][8];
#pragma unroll
  for (int i = 0; i < 4; ++i)
#pragma unroll
    for (int r = 0; r < 8; ++r) m1[i][r] = 0.f;
#pragma unroll
  for (int i = 0; i < 2; ++i)
#pragma unroll
    for (int r = 0; r < 8; ++r) { m2[i][r] = 0.f; rate[i][r] = 0.f; }

  for (int t = 0; t < T; ++t) {
    // ---- layer 0 membranes + spikes -> LDS (thread tid owns column tid)
#pragma unroll 4
    for (int j = 0; j < 64; ++j) {
      size_t g = (size_t)(rowbase + j) * 256 + tid;
      float h  = H0[g];                              // L2-resident
      float mp = (t == 0) ? 0.f : m0g[g];
      float mn = lb0 * mp + h - ((mp > th0) ? th0 : 0.f);
      m0g[g] = mn;
      s0[j * 264 + tid] = (half_t)((mn > th0) ? 1.f : 0.f);
    }
    __syncthreads();

    // ---- layer 1: spikes0[64x256] @ W1^T -> 64x128 (hi/lo split WMMA)
    v8f acc1[4] = {};
#pragma unroll
    for (int kc = 0; kc < 8; ++kc) {
      v16h a = load_a_frag_lds(s0 + tm1 * 16 * 264, 264, kc * 32, lane);
#pragma unroll
      for (int i = 0; i < 4; ++i) {
        v16h bh = load_b_frag(b1hi, tn1b + i, kc, 8, lane);
        v16h bl = load_b_frag(b1lo, tn1b + i, kc, 8, lane);
        acc1[i] = WMMA_F16(a, bh, acc1[i]);
        acc1[i] = WMMA_F16(a, bl, acc1[i]);
      }
    }
#pragma unroll
    for (int i = 0; i < 4; ++i) {
      int n = (tn1b + i) * 16 + (lane & 15);
#pragma unroll
      for (int r = 0; r < 8; ++r) {
        float h  = acc1[i][r] * al1[i] + be1[i];
        float mp = m1[i][r];
        float mn = lb1 * mp + h - ((mp > th1) ? th1 : 0.f);
        m1[i][r] = mn;
        int m = tm1 * 16 + r + ((lane >> 4) << 3);
        s1[m * 136 + n] = (half_t)((mn > th1) ? 1.f : 0.f);
      }
    }
    __syncthreads();

    // ---- layer 2: spikes1[64x128] @ W2^T -> 64x64
    v8f acc2[2] = {};
#pragma unroll
    for (int kc = 0; kc < 4; ++kc) {
      v16h a = load_a_frag_lds(s1 + tm2 * 16 * 136, 136, kc * 32, lane);
#pragma unroll
      for (int i = 0; i < 2; ++i) {
        v16h bh = load_b_frag(b2hi, tn2b + i, kc, 4, lane);
        v16h bl = load_b_frag(b2lo, tn2b + i, kc, 4, lane);
        acc2[i] = WMMA_F16(a, bh, acc2[i]);
        acc2[i] = WMMA_F16(a, bl, acc2[i]);
      }
    }
#pragma unroll
    for (int i = 0; i < 2; ++i) {
      int n = (tn2b + i) * 16 + (lane & 15);
#pragma unroll
      for (int r = 0; r < 8; ++r) {
        float h  = acc2[i][r] * al2[i] + be2[i];
        float mp = m2[i][r];
        float mn = lb2 * mp + h - ((mp > th2) ? th2 : 0.f);
        m2[i][r] = mn;
        float sp = (mn > th2) ? 1.f : 0.f;
        rate[i][r] += sp;
        int m = tm2 * 16 + r + ((lane >> 4) << 3);
        // write-once 128MB stream: keep it out of L2 hot set
        __builtin_nontemporal_store(sp, &out_rec[((size_t)t * 16384 + rowbase + m) * 64 + n]);
      }
    }
    __syncthreads();
  }

  const float invT = 1.0f / (float)T;
#pragma unroll
  for (int i = 0; i < 2; ++i) {
    int n = (tn2b + i) * 16 + (lane & 15);
#pragma unroll
    for (int r = 0; r < 8; ++r) {
      int m = tm2 * 16 + r + ((lane >> 4) << 3);
      out_rate[(size_t)(rowbase + m) * 64 + n] = rate[i][r] * invT;
    }
  }
}

// --------------------------- host launcher ---------------------------------

extern "C" void kernel_launch(void* const* d_in, const int* in_sizes, int n_in,
                              void* d_out, int out_size, void* d_ws, size_t ws_size,
                              hipStream_t stream) {
  (void)in_sizes; (void)n_in; (void)out_size; (void)ws_size;
  const float* x    = (const float*)d_in[0];
  const int*   Tp   = (const int*)d_in[1];
  const float* W0   = (const float*)d_in[2];
  const float* b0   = (const float*)d_in[3];
  const float* g0   = (const float*)d_in[4];
  const float* bb0  = (const float*)d_in[5];
  const float* rm0  = (const float*)d_in[6];
  const float* rv0  = (const float*)d_in[7];
  const float* lb0  = (const float*)d_in[8];
  const float* th0  = (const float*)d_in[9];
  const float* W1   = (const float*)d_in[10];
  const float* b1   = (const float*)d_in[11];
  const float* g1   = (const float*)d_in[12];
  const float* bb1  = (const float*)d_in[13];
  const float* rm1  = (const float*)d_in[14];
  const float* rv1  = (const float*)d_in[15];
  const float* lb1  = (const float*)d_in[16];
  const float* th1  = (const float*)d_in[17];
  const float* W2   = (const float*)d_in[18];
  const float* b2   = (const float*)d_in[19];
  const float* g2   = (const float*)d_in[20];
  const float* bb2  = (const float*)d_in[21];
  const float* rm2  = (const float*)d_in[22];
  const float* rv2  = (const float*)d_in[23];
  const float* lb2  = (const float*)d_in[24];
  const float* th2  = (const float*)d_in[25];

  // Workspace layout (all offsets 32B-aligned; ~33.3 MB total)
  char* ws = (char*)d_ws;
  const size_t MB = 1024ull * 1024ull;
  float*  H0   = (float*)(ws + 0);                     // 16 MB
  float*  m0   = (float*)(ws + 16 * MB);               // 16 MB
  half_t* b0hi = (half_t*)(ws + 32 * MB);              // 512 KB
  half_t* b0lo = (half_t*)(ws + 32 * MB + 512 * 1024);
  half_t* b1hi = (half_t*)(ws + 33 * MB);              // 64 KB
  half_t* b1lo = (half_t*)(ws + 33 * MB + 64 * 1024);
  half_t* b2hi = (half_t*)(ws + 33 * MB + 128 * 1024); // 16 KB
  half_t* b2lo = (half_t*)(ws + 33 * MB + 144 * 1024);
  float*  alpha0 = (float*)(ws + 33 * MB + 160 * 1024);
  float*  beta0  = alpha0 + 256;
  float*  alpha1 = beta0 + 256;
  float*  beta1  = alpha1 + 128;
  float*  alpha2 = beta1 + 128;
  float*  beta2  = alpha2 + 64;

  float* out_rate = (float*)d_out;
  float* out_rec  = out_rate + (size_t)16384 * 64;

  // Prep: pack weights into WMMA B-fragment order (f16 hi/lo), fold BN.
  snn_pack_weights<<<(256 * 1024 + 255) / 256, 256, 0, stream>>>(W0, b0hi, b0lo, 256, 1024);
  snn_pack_weights<<<(128 * 256 + 255) / 256, 256, 0, stream>>>(W1, b1hi, b1lo, 128, 256);
  snn_pack_weights<<<(64 * 128 + 255) / 256, 256, 0, stream>>>(W2, b2hi, b2lo, 64, 128);
  snn_prep_bn<<<1, 256, 0, stream>>>(b0, g0, bb0, rm0, rv0, alpha0, beta0, 256);
  snn_prep_bn<<<1, 128, 0, stream>>>(b1, g1, bb1, rm1, rv1, alpha1, beta1, 128);
  snn_prep_bn<<<1, 64, 0, stream>>>(b2, g2, bb2, rm2, rv2, alpha2, beta2, 64);

  // Stage A: one-shot layer-0 GEMM+BN (x is time-invariant).
  snn_h0_gemm<<<16384 / 32, 256, 0, stream>>>(x, b0hi, b0lo, alpha0, beta0, H0);

  // Stage B: persistent temporal kernel, 64 batch rows per block.
  snn_temporal<<<16384 / 64, 256, 0, stream>>>(H0, m0,
      b1hi, b1lo, b2hi, b2lo, alpha1, beta1, alpha2, beta2,
      lb0, th0, lb1, th1, lb2, th2, Tp, out_rate, out_rec);
}
